// QKVBimodalCSRPool_2233382994385
// MI455X (gfx1250) — matrix operand order repaired and checked by reading, not compiled, not run on hardware
//
#include <hip/hip_runtime.h>
#include <hip/hip_bf16.h>

// ---------------- WMMA helpers (gfx1250, wave32) ----------------
typedef __attribute__((ext_vector_type(16))) _Float16 v16h;
typedef __attribute__((ext_vector_type(2)))  _Float16 v2h;
typedef __attribute__((ext_vector_type(8)))  float    v8f;
typedef int v4i __attribute__((vector_size(16)));

static __device__ __forceinline__ v8f wmma16(v16h a, v16h b, v8f c) {
  // D = A(16x32 f16) * B(32x16 f16) + C(16x16 f32)
  return __builtin_amdgcn_wmma_f32_16x16x32_f16(false, a, false, b, (short)0, c,
                                                false, false);
}

// Swizzled tile layout: a 16x32 (A) or 32x16 (B) f16 tile occupies 512 halfs,
// stored so that lane L's whole fragment is the 16 contiguous halfs at L*16.
// Fragment load is then one 32-byte vector LDS read (2x ds_load_b128).
static __device__ __forceinline__ int a_swz(int r, int kk) {
  int hi   = (kk >> 3) & 1;            // lanes 16-31 hold K 8-15 / 24-31
  int lane = r + (hi << 4);
  int kp   = kk - (hi << 3);           // {0..7} u {16..23}
  int slot = (kp >= 16) ? (kp - 8) : kp;
  return lane * 16 + slot;
}
static __device__ __forceinline__ int b_swz(int kk, int c) {
  int hi   = (kk >> 3) & 1;            // lanes 16-31 hold K 8-15 / 24-31
  int lane = c + (hi << 4);
  int kp   = kk - (hi << 3);           // {0..7} u {16..23}
  int slot = (kp >= 16) ? (kp - 8) : kp;
  return lane * 16 + slot;
}

static __device__ __forceinline__ v16h frag_ld(const _Float16* tile) {
  return *(const v16h*)(tile + (threadIdx.x & 31) * 16);
}

// ---- async global->LDS staging (CDNA5 GLOBAL_LOAD_ASYNC_TO_LDS_B128) ----
#if defined(__gfx1250__) && __has_builtin(__builtin_amdgcn_global_load_async_to_lds_b128)
#define HAVE_ASYNC_LDS 1
#else
#define HAVE_ASYNC_LDS 0
#endif

static __device__ __forceinline__ void stage_weights(const _Float16* src,
                                                     _Float16* dst, int halfs,
                                                     int tid, int nthreads) {
  int n16 = halfs / 8;  // 16-byte chunks
  for (int i = tid; i < n16; i += nthreads) {
#if HAVE_ASYNC_LDS
    __builtin_amdgcn_global_load_async_to_lds_b128(
        (__attribute__((address_space(1))) v4i*)((const uint4*)src + i),
        (__attribute__((address_space(3))) v4i*)((uint4*)dst + i),
        /*offset=*/0, /*cpol=*/0);
#else
    ((uint4*)dst)[i] = ((const uint4*)src)[i];
#endif
  }
}
static __device__ __forceinline__ void wait_async0() {
#if HAVE_ASYNC_LDS
#if __has_builtin(__builtin_amdgcn_s_wait_asynccnt)
  __builtin_amdgcn_s_wait_asynccnt(0);
#else
  asm volatile("s_wait_asynccnt 0" ::: "memory");
#endif
#endif
}

// Monotone order-preserving float<->u32 encode for atomic max of signed floats.
static __device__ __forceinline__ unsigned enc_f32(float f) {
  unsigned u = __float_as_uint(f);
  return (u & 0x80000000u) ? ~u : (u | 0x80000000u);
}
static __device__ __forceinline__ float dec_f32(unsigned u) {
  return __uint_as_float((u & 0x80000000u) ? (u ^ 0x80000000u) : ~u);
}

// ---------------- utility kernels ----------------
__global__ void k_zero(unsigned* p, int n) {
  int i = blockIdx.x * blockDim.x + threadIdx.x;
  if (i < n) p[i] = 0u;
}

__global__ void k_dense_idx(const int* __restrict__ csr, int* __restrict__ didx,
                            int N, int V) {
  int v = blockIdx.x * blockDim.x + threadIdx.x;
  if (v >= V) return;
  int lo = 0, hi = N - 1;
  while (lo < hi) {               // largest s with csr[s] <= v
    int mid = (lo + hi + 1) >> 1;
    if (csr[mid] <= v) lo = mid; else hi = mid - 1;
  }
  didx[v] = lo;
}

// Wc = W_main2(32x32) @ WQ(32x8)  -- fold since no ReLU between them
__global__ void k_fold(const float* __restrict__ W2, const float* __restrict__ WQ,
                       float* __restrict__ Wc) {
  int i = threadIdx.x;
  if (i >= 32 * 8) return;
  int r = i >> 3, c = i & 7;
  float s = 0.f;
#pragma unroll
  for (int k = 0; k < 32; ++k) s += W2[r * 32 + k] * WQ[k * 8 + c];
  Wc[r * 8 + c] = s;
}

// Convert f32 weight [K][C] row-major into swizzled f16 B-tiles, padded to Kp x Cp.
__global__ void k_prep_b(const float* __restrict__ W, _Float16* __restrict__ out,
                         int K, int C, int Kp, int Cp) {
  int i = blockIdx.x * blockDim.x + threadIdx.x;
  if (i >= Kp * Cp) return;
  int r = i / Cp, c = i % Cp;
  float v = (r < K && c < C) ? W[r * C + c] : 0.f;
  int tile = (r >> 5) * (Cp >> 4) + (c >> 4);
  out[tile * 512 + b_swz(r & 31, c & 15)] = (_Float16)v;
}

// ---------------- WMMA kernels ----------------
// qN = relu(x_main @ W_main1) @ Wc + bQ          [N,8]
#define QWAVES 4
__global__ void k_queries(const float* __restrict__ x_main,
                          const _Float16* __restrict__ w1s,
                          const _Float16* __restrict__ wcs,
                          const float* __restrict__ bQ,
                          float* __restrict__ qN, int N) {
  __shared__ __align__(32) _Float16 sW1[4 * 512];
  __shared__ __align__(32) _Float16 sWc[512];
  __shared__ __align__(32) _Float16 sX[QWAVES][2 * 512];
  __shared__ __align__(32) _Float16 sH[QWAVES][512];
  int tid = threadIdx.x, lane = tid & 31, wave = tid >> 5;
  stage_weights(w1s, sW1, 4 * 512, tid, blockDim.x);
  stage_weights(wcs, sWc, 512, tid, blockDim.x);
  int base = (blockIdx.x * QWAVES + wave) * 16;
  for (int i = lane; i < 16 * 64; i += 32) {
    int r = i >> 6, c = i & 63, g = base + r;
    float v = (g < N) ? x_main[g * 64 + c] : 0.f;
    sX[wave][(c >> 5) * 512 + a_swz(r, c & 31)] = (_Float16)v;
  }
  wait_async0();
  __syncthreads();
  bool active = base < N;
  if (active) {
    v16h a0 = frag_ld(&sX[wave][0]);
    v16h a1 = frag_ld(&sX[wave][512]);
    v16h b00 = frag_ld(&sW1[0 * 512]), b01 = frag_ld(&sW1[1 * 512]);
    v16h b10 = frag_ld(&sW1[2 * 512]), b11 = frag_ld(&sW1[3 * 512]);
    v8f c0 = {}, c1 = {};
    c0 = wmma16(a0, b00, c0);
    c0 = wmma16(a1, b10, c0);
    c1 = wmma16(a0, b01, c1);
    c1 = wmma16(a1, b11, c1);
    int col = lane & 15, mb = (lane >> 4) << 3;
#pragma unroll
    for (int r = 0; r < 8; ++r) {
      sH[wave][a_swz(mb + r, col)]      = (_Float16)fmaxf(c0[r], 0.f);
      sH[wave][a_swz(mb + r, col + 16)] = (_Float16)fmaxf(c1[r], 0.f);
    }
  }
  __syncthreads();
  if (active) {
    v8f c = {};
    c = wmma16(frag_ld(&sH[wave][0]), frag_ld(sWc), c);
    int col = lane & 15, mb = (lane >> 4) << 3;
    if (col < 8) {
      float bq = bQ[col];
#pragma unroll
      for (int r = 0; r < 8; ++r) {
        int g = base + mb + r;
        if (g < N) qN[g * 8 + col] = c[r] + bq;
      }
    }
  }
}

// h = relu(x_map @ W_map1)  [V,32] (f16) ; set_ctx = segment_max(h) via u32 atomicMax
#define HWAVES 4
__global__ void k_h(const float* __restrict__ x_map,
                    const _Float16* __restrict__ w1s,
                    const int* __restrict__ didx, _Float16* __restrict__ h16,
                    unsigned* __restrict__ setctx, int V) {
  __shared__ __align__(32) _Float16 sW[2 * 512];
  __shared__ __align__(32) _Float16 sX[HWAVES][512];
  int tid = threadIdx.x, lane = tid & 31, wave = tid >> 5;
  stage_weights(w1s, sW, 2 * 512, tid, blockDim.x);
  int base = (blockIdx.x * HWAVES + wave) * 16;
  for (int i = lane; i < 16 * 32; i += 32) {
    int r = i >> 5, c = i & 31, g = base + r;
    float v = (g < V && c < 16) ? x_map[g * 16 + c] : 0.f;   // K padded 16->32
    sX[wave][a_swz(r, c)] = (_Float16)v;
  }
  wait_async0();
  __syncthreads();
  if (base >= V) return;
  v16h a = frag_ld(&sX[wave][0]);
  v16h b0 = frag_ld(&sW[0]), b1 = frag_ld(&sW[512]);
  v8f c0 = {}, c1 = {};
  c0 = wmma16(a, b0, c0);
  c1 = wmma16(a, b1, c1);
  int col = lane & 15, mb = (lane >> 4) << 3;
#pragma unroll
  for (int r = 0; r < 8; ++r) {
    int g = base + mb + r;
    if (g < V) {
      float v0 = fmaxf(c0[r], 0.f), v1 = fmaxf(c1[r], 0.f);
      h16[g * 32 + col]      = (_Float16)v0;
      h16[g * 32 + col + 16] = (_Float16)v1;
      int s = didx[g];
      atomicMax(&setctx[s * 32 + col],      __float_as_uint(v0)); // relu>=0
      atomicMax(&setctx[s * 32 + col + 16], __float_as_uint(v1));
    }
  }
}

// h_map = relu(cat(h, set_ctx[seg]) @ W_map2); keys = h_map@WK + bK;
// comp = dot(keys, qN[seg]) / sqrt(8); atomic segment max of comp.
#define KWAVES 4
__global__ void k_keys(const _Float16* __restrict__ h16,
                       const unsigned* __restrict__ setctx,
                       const int* __restrict__ didx,
                       const _Float16* __restrict__ w2s,
                       const _Float16* __restrict__ wks,
                       const float* __restrict__ bK, const float* __restrict__ qN,
                       float* __restrict__ compv, unsigned* __restrict__ segmax,
                       int V) {
  __shared__ __align__(32) _Float16 sW2[4 * 512];
  __shared__ __align__(32) _Float16 sWK[512];
  __shared__ __align__(32) _Float16 sC[KWAVES][2 * 512];
  __shared__ __align__(32) _Float16 sM[KWAVES][512];
  __shared__ float sK[KWAVES][16 * 8];
  int tid = threadIdx.x, lane = tid & 31, wave = tid >> 5;
  stage_weights(w2s, sW2, 4 * 512, tid, blockDim.x);
  stage_weights(wks, sWK, 512, tid, blockDim.x);
  int base = (blockIdx.x * KWAVES + wave) * 16;
  for (int i = lane; i < 16 * 64; i += 32) {
    int r = i >> 6, c = i & 63, g = base + r;
    _Float16 val = (_Float16)0.f;
    if (g < V) {
      if (c < 32) val = h16[g * 32 + c];
      else        val = (_Float16)__uint_as_float(setctx[didx[g] * 32 + (c - 32)]);
    }
    sC[wave][(c >> 5) * 512 + a_swz(r, c & 31)] = val;
  }
  wait_async0();
  __syncthreads();
  bool active = base < V;
  if (active) {
    v16h a0 = frag_ld(&sC[wave][0]);
    v16h a1 = frag_ld(&sC[wave][512]);
    v16h b00 = frag_ld(&sW2[0 * 512]), b01 = frag_ld(&sW2[1 * 512]);
    v16h b10 = frag_ld(&sW2[2 * 512]), b11 = frag_ld(&sW2[3 * 512]);
    v8f c0 = {}, c1 = {};
    c0 = wmma16(a0, b00, c0);
    c0 = wmma16(a1, b10, c0);
    c1 = wmma16(a0, b01, c1);
    c1 = wmma16(a1, b11, c1);
    int col = lane & 15, mb = (lane >> 4) << 3;
#pragma unroll
    for (int r = 0; r < 8; ++r) {
      sM[wave][a_swz(mb + r, col)]      = (_Float16)fmaxf(c0[r], 0.f);
      sM[wave][a_swz(mb + r, col + 16)] = (_Float16)fmaxf(c1[r], 0.f);
    }
  }
  __syncthreads();
  if (active) {
    v8f c = {};
    c = wmma16(frag_ld(&sM[wave][0]), frag_ld(sWK), c);
    int col = lane & 15, mb = (lane >> 4) << 3;
    if (col < 8) {
      float bk = bK[col];
#pragma unroll
      for (int r = 0; r < 8; ++r) sK[wave][(mb + r) * 8 + col] = c[r] + bk;
    }
  }
  __syncthreads();
  if (active && lane < 16) {
    int g = base + lane;
    if (g < V) {
      int s = didx[g];
      float acc = 0.f;
#pragma unroll
      for (int j = 0; j < 8; ++j) acc += sK[wave][lane * 8 + j] * qN[s * 8 + j];
      acc *= 0.35355339059327373f;   // 1/sqrt(8)
      compv[g] = acc;
      atomicMax(&segmax[s], enc_f32(acc));
    }
  }
}

__global__ void k_cexp(const float* __restrict__ compv, const int* __restrict__ didx,
                       const unsigned* __restrict__ segmax,
                       float* __restrict__ cexpv, float* __restrict__ segsum, int V) {
  int v = blockIdx.x * blockDim.x + threadIdx.x;
  if (v >= V) return;
  int s = didx[v];
  float e = __expf(compv[v] - dec_f32(segmax[s]));
  cexpv[v] = e;
  atomicAdd(&segsum[s], e);
}

// t = relu(x_mod @ W_mod1)  [V,128] (f16)   -- dominant GEMM, 32 WMMAs / wave tile
#define TWAVES 8
__global__ void k_t(const float* __restrict__ x_mod,
                    const _Float16* __restrict__ w1s,  // 32 tiles (kc*8+nc)
                    _Float16* __restrict__ t16, int V) {
  __shared__ __align__(32) _Float16 sW[32 * 512];
  __shared__ __align__(32) _Float16 sX[TWAVES][4 * 512];
  int tid = threadIdx.x, lane = tid & 31, wave = tid >> 5;
  stage_weights(w1s, sW, 32 * 512, tid, blockDim.x);
  int base = (blockIdx.x * TWAVES + wave) * 16;
  for (int i = lane; i < 16 * 128; i += 32) {
    int r = i >> 7, c = i & 127, g = base + r;
    float v = (g < V) ? x_mod[g * 128 + c] : 0.f;
    sX[wave][(c >> 5) * 512 + a_swz(r, c & 31)] = (_Float16)v;
  }
  wait_async0();
  __syncthreads();
  if (base >= V) return;
  v8f zero = {};
  v8f acc[8];
#pragma unroll
  for (int n = 0; n < 8; ++n) acc[n] = zero;
#pragma unroll
  for (int kc = 0; kc < 4; ++kc) {
    v16h a = frag_ld(&sX[wave][kc * 512]);
    v16h b[8];
#pragma unroll
    for (int n = 0; n < 8; ++n) b[n] = frag_ld(&sW[(kc * 8 + n) * 512]);
#pragma unroll
    for (int n = 0; n < 8; ++n) acc[n] = wmma16(a, b[n], acc[n]);
  }
  int col = lane & 15, mb = (lane >> 4) << 3;
#pragma unroll
  for (int n = 0; n < 8; ++n)
#pragma unroll
    for (int r = 0; r < 8; ++r) {
      int g = base + mb + r;
      if (g < V) t16[(size_t)g * 128 + n * 16 + col] = (_Float16)fmaxf(acc[n][r], 0.f);
    }
}

// p[s] = sum_v att[v] * t[v]  (segments contiguous in CSR; 1 block / segment)
__global__ void k_pool(const int* __restrict__ csr, const float* __restrict__ cexpv,
                       const _Float16* __restrict__ t16,
                       const float* __restrict__ segsum, float* __restrict__ p) {
  int s = blockIdx.x, tid = threadIdx.x;  // 64 threads, 2 channels each
  int b = csr[s], e = csr[s + 1];
  float denom = segsum[s] + 1e-12f;
  const v2h* t2 = (const v2h*)t16;
  float a0 = 0.f, a1 = 0.f;
  for (int v = b; v < e; ++v) {
    float att = cexpv[v] / denom;
    v2h w = t2[(size_t)v * 64 + tid];
    a0 += att * (float)w[0];
    a1 += att * (float)w[1];
  }
  p[s * 128 + 2 * tid]     = a0;
  p[s * 128 + 2 * tid + 1] = a1;
}

// x_pool = (p @ W_mod2) * gate ; x_seen
#define FWAVES 8
__global__ void k_final(const float* __restrict__ p,
                        const _Float16* __restrict__ w2s,  // 32 tiles
                        const int* __restrict__ csr, const unsigned* __restrict__ segmax,
                        const float* __restrict__ g_w, const float* __restrict__ g_b,
                        float* __restrict__ out_pool, float* __restrict__ out_seen,
                        int N) {
  __shared__ __align__(32) _Float16 sW[32 * 512];
  __shared__ __align__(32) _Float16 sP[FWAVES][4 * 512];
  int tid = threadIdx.x, lane = tid & 31, wave = tid >> 5;
  stage_weights(w2s, sW, 32 * 512, tid, blockDim.x);
  int base = (blockIdx.x * FWAVES + wave) * 16;
  for (int i = lane; i < 16 * 128; i += 32) {
    int r = i >> 7, c = i & 127, g = base + r;
    float v = (g < N) ? p[g * 128 + c] : 0.f;
    sP[wave][(c >> 5) * 512 + a_swz(r, c & 31)] = (_Float16)v;
  }
  wait_async0();
  __syncthreads();
  if (base >= N) return;
  v8f zero = {};
  v8f acc[8];
#pragma unroll
  for (int n = 0; n < 8; ++n) acc[n] = zero;
#pragma unroll
  for (int kc = 0; kc < 4; ++kc) {
    v16h a = frag_ld(&sP[wave][kc * 512]);
    v16h b[8];
#pragma unroll
    for (int n = 0; n < 8; ++n) b[n] = frag_ld(&sW[(kc * 8 + n) * 512]);
#pragma unroll
    for (int n = 0; n < 8; ++n) acc[n] = wmma16(a, b[n], acc[n]);
  }
  int col = lane & 15, mb = (lane >> 4) << 3;
  float gw = g_w[0], gb = g_b[0];
  float gate[8];
#pragma unroll
  for (int r = 0; r < 8; ++r) {
    int g = base + mb + r;
    float gv = 0.f;
    if (g < N) {
      int cnt = csr[g + 1] - csr[g];
      float sm = (cnt > 0) ? dec_f32(segmax[g]) : 0.f;
      gv = tanhf(fmaxf(gw * sm + gb, 0.f));
      if (col == 0) out_seen[g] = (cnt > 0) ? 1.f : 0.f;
    }
    gate[r] = gv;
  }
#pragma unroll
  for (int n = 0; n < 8; ++n)
#pragma unroll
    for (int r = 0; r < 8; ++r) {
      int g = base + mb + r;
      if (g < N) out_pool[(size_t)g * 128 + n * 16 + col] = acc[n][r] * gate[r];
    }
}

// ---------------- launch ----------------
extern "C" void kernel_launch(void* const* d_in, const int* in_sizes, int n_in,
                              void* d_out, int out_size, void* d_ws, size_t ws_size,
                              hipStream_t stream) {
  (void)n_in; (void)out_size; (void)ws_size;
  const float* x_main  = (const float*)d_in[0];
  const float* x_mod   = (const float*)d_in[1];
  const float* x_map   = (const float*)d_in[2];
  const int*   csr     = (const int*)d_in[3];
  const float* W_main1 = (const float*)d_in[4];
  const float* W_main2 = (const float*)d_in[5];
  const float* W_map1  = (const float*)d_in[6];
  const float* W_map2  = (const float*)d_in[7];
  const float* W_mod1  = (const float*)d_in[8];
  const float* W_mod2  = (const float*)d_in[9];
  const float* WQ      = (const float*)d_in[10];
  const float* bQ      = (const float*)d_in[11];
  const float* WK      = (const float*)d_in[12];
  const float* bK      = (const float*)d_in[13];
  const float* g_w     = (const float*)d_in[14];
  const float* g_b     = (const float*)d_in[15];

  const int N = in_sizes[0] / 64;    // x_main [N,64]
  const int V = in_sizes[2] / 16;    // x_map  [V,16]

  char* ws = (char*)d_ws;
  size_t off = 0;
  auto alloc = [&](size_t bytes) -> void* {
    void* ptr = ws + off;
    off = (off + bytes + 255) & ~(size_t)255;
    return ptr;
  };
  int*      didx   = (int*)alloc((size_t)V * 4);
  float*    Wc     = (float*)alloc(32 * 8 * 4);
  float*    qN     = (float*)alloc((size_t)N * 8 * 4);
  _Float16* h16    = (_Float16*)alloc((size_t)V * 32 * 2);
  unsigned* setctx = (unsigned*)alloc((size_t)N * 32 * 4);
  float*    compv  = (float*)alloc((size_t)V * 4);
  float*    cexpv  = (float*)alloc((size_t)V * 4);
  unsigned* segmax = (unsigned*)alloc((size_t)N * 4);
  float*    segsum = (float*)alloc((size_t)N * 4);
  _Float16* t16    = (_Float16*)alloc((size_t)V * 128 * 2);
  float*    pbuf   = (float*)alloc((size_t)N * 128 * 4);
  // swizzled f16 weight images
  _Float16* w_main1_s = (_Float16*)alloc(64 * 32 * 2);
  _Float16* wc_s      = (_Float16*)alloc(32 * 16 * 2);
  _Float16* w_map1_s  = (_Float16*)alloc(32 * 32 * 2);   // K padded 16->32
  _Float16* w_map2_s  = (_Float16*)alloc(64 * 32 * 2);
  _Float16* wk_s      = (_Float16*)alloc(32 * 16 * 2);
  _Float16* w_mod1_s  = (_Float16*)alloc(128 * 128 * 2);
  _Float16* w_mod2_s  = (_Float16*)alloc(128 * 128 * 2);

  float* out_pool = (float*)d_out;
  float* out_seen = out_pool + (size_t)N * 128;

  // deterministic re-init of atomic accumulators every call
  k_zero<<<(N * 32 + 255) / 256, 256, 0, stream>>>(setctx, N * 32);
  k_zero<<<(N + 255) / 256, 256, 0, stream>>>(segmax, N);
  k_zero<<<(N + 255) / 256, 256, 0, stream>>>((unsigned*)segsum, N);

  k_dense_idx<<<(V + 255) / 256, 256, 0, stream>>>(csr, didx, N, V);
  k_fold<<<1, 256, 0, stream>>>(W_main2, WQ, Wc);

  auto prep = [&](const float* W, _Float16* o, int K, int C, int Kp, int Cp) {
    int tot = Kp * Cp;
    k_prep_b<<<(tot + 255) / 256, 256, 0, stream>>>(W, o, K, C, Kp, Cp);
  };
  prep(W_main1, w_main1_s, 64, 32, 64, 32);
  prep(Wc,      wc_s,      32, 8,  32, 16);
  prep(W_map1,  w_map1_s,  16, 32, 32, 32);
  prep(W_map2,  w_map2_s,  64, 32, 64, 32);
  prep(WK,      wk_s,      32, 8,  32, 16);
  prep(W_mod1,  w_mod1_s, 128, 128, 128, 128);
  prep(W_mod2,  w_mod2_s, 128, 128, 128, 128);

  int qtiles = (N + 15) / 16;
  int vtiles = (V + 15) / 16;
  k_queries<<<(qtiles + QWAVES - 1) / QWAVES, QWAVES * 32, 0, stream>>>(
      x_main, w_main1_s, wc_s, bQ, qN, N);
  k_h<<<(vtiles + HWAVES - 1) / HWAVES, HWAVES * 32, 0, stream>>>(
      x_map, w_map1_s, didx, h16, setctx, V);
  k_keys<<<(vtiles + KWAVES - 1) / KWAVES, KWAVES * 32, 0, stream>>>(
      h16, setctx, didx, w_map2_s, wk_s, bK, qN, compv, segmax, V);
  k_cexp<<<(V + 255) / 256, 256, 0, stream>>>(compv, didx, segmax, cexpv, segsum, V);
  k_t<<<(vtiles + TWAVES - 1) / TWAVES, TWAVES * 32, 0, stream>>>(x_mod, w_mod1_s, t16, V);
  k_pool<<<N, 64, 0, stream>>>(csr, cexpv, t16, segsum, pbuf);
  k_final<<<(qtiles + FWAVES - 1) / FWAVES, FWAVES * 32, 0, stream>>>(
      pbuf, w_mod2_s, csr, segmax, g_w, g_b, out_pool, out_seen, N);
}
